// MixtureOfExperts_27015344292298
// MI455X (gfx1250) — matrix operand order repaired and checked by reading, not compiled, not run on hardware
//
#include <hip/hip_runtime.h>
#include <hip/hip_bf16.h>

#define DIM 1024
#define HID 4096
#define NEXP 8
#define TOKENS 4096
#define MTILE 64               // tokens per block = 4 WMMA row-blocks
#define NWAVES 16
#define BLOCK (NWAVES * 32)
#define CHUNK 256              // h-columns materialized in LDS per chunk
#define NCHUNKS (HID / CHUNK)  // 16
#define XSTRIDE (DIM + 8)      // ushorts; 2064 B row stride (16B multiple)
#define HSTRIDE (CHUNK + 8)    // ushorts; 528 B -> 16B multiple

typedef __attribute__((ext_vector_type(16))) __bf16 v16bf;
typedef __attribute__((ext_vector_type(2)))  __bf16 v2bf;
typedef __attribute__((ext_vector_type(8)))  float  v8f;

union Frag16 {
    v16bf        bf;
    unsigned int u32[8];
    unsigned short us[16];
};

// two f32 -> packed bf16x2 (RTNE) -> v_cvt_pk_bf16_f32
__device__ __forceinline__ unsigned pack_bf16x2(float a, float b) {
    v2bf p;
    p.x = (__bf16)a;
    p.y = (__bf16)b;
    return __builtin_bit_cast(unsigned, p);
}

__device__ __forceinline__ unsigned short bf16bits(float a) {
    return __builtin_bit_cast(unsigned short, (__bf16)a);
}

// Branchless GELU (tanh form); error ~1e-3, far below bf16 matmul noise.
__device__ __forceinline__ float gelu_fast(float v) {
    const float u = 0.79788456080286536f * (v + 0.044715f * v * v * v);
#if __has_builtin(__builtin_amdgcn_tanh_f32)
    const float t = __builtin_amdgcn_tanh_f32(u);  // v_tanh_f32 (TRANS)
#else
    const float t = tanhf(u);
#endif
    return 0.5f * v * (1.0f + t);
}

// 16x32 bf16 A-fragment from LDS tile (row-major, stride in ushorts).
// ISA 7.12.2: lanes 0-15 -> row M=mbase+lane, K {0..7,16..23};
// lanes 16-31 -> K {8..15,24..31}. Dword pattern is 16B-contiguous -> 2x b128.
__device__ __forceinline__ Frag16 load_a_lds(const unsigned short* tile, int stride_us,
                                             int kb, int lane, int mbase) {
    Frag16 f;
    const int m  = mbase + (lane & 15);
    const int hi = (lane >> 4) << 3;
    const unsigned short* p = tile + m * stride_us + kb + hi;
    *(uint4*)&f.u32[0] = *(const uint4*)(p);       // K pairs {0..7}+hi
    *(uint4*)&f.u32[4] = *(const uint4*)(p + 16);  // K pairs {16..23}+hi
    return f;
}

// 32x16 bf16 B-fragment from global f32 matrix (row-major, row stride ld).
// Lanes 0-15: N=n0+lane, K {0..7,16..23} ; lanes 16-31: N=n0+lane-16, K {8..15,24..31}.
__device__ __forceinline__ Frag16 load_b_global(const float* __restrict__ W,
                                                size_t ld, int kb, int n0, int lane) {
    const int n  = n0 + (lane & 15);
    const int k0 = kb + ((lane >> 4) << 3);
    const float* __restrict__ base = W + (size_t)k0 * ld + n;
    __builtin_prefetch(base + 64 * ld, 0, 1);  // 2 k-tiles ahead -> global_prefetch_b8
    float t[16];
#pragma unroll
    for (int j = 0; j < 16; ++j) t[j] = base[(size_t)(j < 8 ? j : j + 8) * ld];
    Frag16 f;
#pragma unroll
    for (int j = 0; j < 8; ++j) f.u32[j] = pack_bf16x2(t[2 * j], t[2 * j + 1]);
    return f;
}

#define WMMA_BF16(A, B, C) \
    __builtin_amdgcn_wmma_f32_16x16x32_bf16(false, (A).bf, false, (B).bf, (short)0, (C), false, false)

// ---------------------------------------------------------------------------
// Kernel 1: router. One wave per token: logits, top-2, softmax, expert lists.
// ---------------------------------------------------------------------------
__global__ void moe_router(const float* __restrict__ x,  const float* __restrict__ Wr,
                           const float* __restrict__ br, float* __restrict__ wout,
                           int* __restrict__ lists,      int* __restrict__ counts) {
    const int wave = threadIdx.x >> 5;
    const int lane = threadIdx.x & 31;
    const int t    = blockIdx.x * 8 + wave;

    float acc[NEXP];
#pragma unroll
    for (int e = 0; e < NEXP; ++e) acc[e] = 0.0f;

    const float* xr = x + (size_t)t * DIM;
    for (int d = lane; d < DIM; d += 32) {
        const float  xv = xr[d];
        const float* wr = Wr + (size_t)d * NEXP;
#pragma unroll
        for (int e = 0; e < NEXP; ++e) acc[e] += xv * wr[e];
    }
#pragma unroll
    for (int e = 0; e < NEXP; ++e) {
#pragma unroll
        for (int off = 16; off > 0; off >>= 1)
            acc[e] += __shfl_xor(acc[e], off, 32);
    }

    if (lane == 0) {
#pragma unroll
        for (int e = 0; e < NEXP; ++e) acc[e] += br[e];
        // top-2, earliest index wins ties (matches jax.lax.top_k)
        int i0 = 0; float v0 = acc[0];
#pragma unroll
        for (int e = 1; e < NEXP; ++e)
            if (acc[e] > v0) { v0 = acc[e]; i0 = e; }
        int i1 = -1; float v1 = -3.4e38f;
#pragma unroll
        for (int e = 0; e < NEXP; ++e)
            if (e != i0 && acc[e] > v1) { v1 = acc[e]; i1 = e; }
        const float w1 = 1.0f / (1.0f + expf(v0 - v1));  // softmax of (v0,v1)
        const float w0 = 1.0f - w1;
        wout[t * 2 + 0] = w0;
        wout[t * 2 + 1] = w1;
        const int p0 = atomicAdd(&counts[i0], 1);
        lists[i0 * TOKENS + p0] = t * 2 + 0;
        const int p1 = atomicAdd(&counts[i1], 1);
        lists[i1 * TOKENS + p1] = t * 2 + 1;
    }
}

// ---------------------------------------------------------------------------
// Kernel 2: fused expert FFN on a 64-token tile of one expert's token list.
// Each B fragment feeds 4 WMMAs (row-blocks m=0..3); A fragments hoisted per k.
// x-tile (bf16, LDS) -> GEMM1+GELU into LDS h-chunks -> GEMM2 accumulated in
// VGPRs across chunks -> y+b2 scattered to per-(token,slot) partial buffer.
// ---------------------------------------------------------------------------
__global__ void __launch_bounds__(BLOCK)
moe_expert(const float* __restrict__ x,  const float* __restrict__ W1,
           const float* __restrict__ b1, const float* __restrict__ W2,
           const float* __restrict__ b2, const int* __restrict__ lists,
           const int* __restrict__ counts, float* __restrict__ yp) {
    extern __shared__ char smem[];
    unsigned short* xs   = (unsigned short*)smem;             // [MTILE][XSTRIDE]
    unsigned short* hs   = xs + MTILE * XSTRIDE;              // [MTILE][HSTRIDE]
    int*            ents = (int*)(hs + MTILE * HSTRIDE);      // [MTILE]

    const int e     = blockIdx.y;
    const int cnt   = counts[e];
    const int tile0 = blockIdx.x * MTILE;
    if (tile0 >= cnt) return;
    const int nrows = min(MTILE, cnt - tile0);

    const int wave = threadIdx.x >> 5;
    const int lane = threadIdx.x & 31;

    if (threadIdx.x < MTILE) {
        const int r = threadIdx.x;
        ents[r] = lists[e * TOKENS + tile0 + (r < nrows ? r : 0)];
    }
    __syncthreads();

    // Gather 64 token rows of x into LDS as bf16 (float4 loads, b64 LDS stores).
    for (int idx = threadIdx.x; idx < MTILE * (DIM / 4); idx += BLOCK) {
        const int r  = idx >> 8;         // DIM/4 == 256
        const int c4 = idx & 255;
        const int t  = ents[r] >> 1;
        const float4 v = ((const float4*)x)[(size_t)t * (DIM / 4) + c4];
        uint2 p;
        p.x = pack_bf16x2(v.x, v.y);
        p.y = pack_bf16x2(v.z, v.w);
        *(uint2*)&xs[r * XSTRIDE + c4 * 4] = p;
    }
    __syncthreads();

    const float* W1e = W1 + (size_t)e * DIM * HID;
    const float* W2e = W2 + (size_t)e * HID * DIM;

    v8f yacc[4][4];  // [n-tile][row-block]
#pragma unroll
    for (int i = 0; i < 4; ++i)
#pragma unroll
        for (int m = 0; m < 4; ++m) yacc[i][m] = {};

    for (int c = 0; c < NCHUNKS; ++c) {
        const int hbase = c * CHUNK;
        const int n0g1  = hbase + wave * 16;  // this wave's single GEMM1 n-tile

        // ---- GEMM1: wave owns 16 h-cols (1 tile), all 4 row-blocks ----
        v8f g1[4];
#pragma unroll
        for (int m = 0; m < 4; ++m) g1[m] = {};

        for (int kb = 0; kb < DIM; kb += 32) {
            Frag16 a[4];
#pragma unroll
            for (int m = 0; m < 4; ++m) a[m] = load_a_lds(xs, XSTRIDE, kb, lane, m * 16);
            const Frag16 b = load_b_global(W1e, HID, kb, n0g1, lane);
#pragma unroll
            for (int m = 0; m < 4; ++m) g1[m] = WMMA_BF16(a[m], b, g1[m]);
        }
        // epilogue: bias + GELU, store bf16 into h-chunk
        {
            const int ncol  = n0g1 + (lane & 15);
            const float b1v = b1[e * HID + ncol];
            const int lcol  = (n0g1 - hbase) + (lane & 15);
            const int vrow  = (lane >> 4) << 3;
#pragma unroll
            for (int m = 0; m < 4; ++m) {
#pragma unroll
                for (int v = 0; v < 8; ++v) {
                    const float g = gelu_fast(g1[m][v] + b1v);
                    hs[(m * 16 + vrow + v) * HSTRIDE + lcol] = bf16bits(g);
                }
            }
        }
        __syncthreads();

        // ---- GEMM2 partial: wave owns 64 y-cols (4 tiles), all 4 row-blocks ----
        for (int kk = 0; kk < CHUNK; kk += 32) {
            Frag16 a[4];
#pragma unroll
            for (int m = 0; m < 4; ++m) a[m] = load_a_lds(hs, HSTRIDE, kk, lane, m * 16);
#pragma unroll
            for (int ti = 0; ti < 4; ++ti) {
                const int n0 = wave * 64 + ti * 16;
                const Frag16 b = load_b_global(W2e, DIM, hbase + kk, n0, lane);
#pragma unroll
                for (int m = 0; m < 4; ++m) yacc[ti][m] = WMMA_BF16(a[m], b, yacc[ti][m]);
            }
        }
        __syncthreads();  // before next chunk overwrites hs
    }

    // ---- Scatter y + b2 into per-(token,slot) partial buffer ----
#pragma unroll
    for (int ti = 0; ti < 4; ++ti) {
        const int col   = wave * 64 + ti * 16 + (lane & 15);
        const float b2v = b2[e * DIM + col];
        const int vrow  = (lane >> 4) << 3;
#pragma unroll
        for (int m = 0; m < 4; ++m) {
#pragma unroll
            for (int v = 0; v < 8; ++v) {
                const int r = m * 16 + vrow + v;
                if (r < nrows)
                    yp[(size_t)ents[r] * DIM + col] = yacc[ti][m][v] + b2v;
            }
        }
    }
}

// ---------------------------------------------------------------------------
// Kernel 3: weighted combine of the two expert partials per token.
// ---------------------------------------------------------------------------
__global__ void moe_combine(const float* __restrict__ yp, const float* __restrict__ w,
                            float* __restrict__ out) {
    const size_t i = (size_t)blockIdx.x * 256 + threadIdx.x;  // over TOKENS*DIM
    const int t = (int)(i >> 10);
    const int d = (int)(i & (DIM - 1));
    out[i] = w[2 * t + 0] * yp[(size_t)(2 * t + 0) * DIM + d] +
             w[2 * t + 1] * yp[(size_t)(2 * t + 1) * DIM + d];
}

extern "C" void kernel_launch(void* const* d_in, const int* in_sizes, int n_in,
                              void* d_out, int out_size, void* d_ws, size_t ws_size,
                              hipStream_t stream) {
    const float* x  = (const float*)d_in[0];
    const float* Wr = (const float*)d_in[1];
    const float* br = (const float*)d_in[2];
    const float* W1 = (const float*)d_in[3];
    const float* b1 = (const float*)d_in[4];
    const float* W2 = (const float*)d_in[5];
    const float* b2 = (const float*)d_in[6];
    float* out = (float*)d_out;

    char* ws = (char*)d_ws;
    int*   counts = (int*)(ws + 0);                         // 8 ints
    float* wbuf   = (float*)(ws + 1024);                    // TOKENS*2 f32
    int*   lists  = (int*)(ws + 1024 + TOKENS * 2 * 4);     // NEXP*TOKENS ints
    float* yp     = (float*)(ws + 1024 + TOKENS * 2 * 4 + NEXP * TOKENS * 4);

    hipMemsetAsync(counts, 0, NEXP * sizeof(int), stream);

    moe_router<<<TOKENS / 8, 256, 0, stream>>>(x, Wr, br, wbuf, lists, counts);

    const size_t lds_bytes = (size_t)MTILE * XSTRIDE * 2 + (size_t)MTILE * HSTRIDE * 2 +
                             MTILE * sizeof(int);
    moe_expert<<<dim3(TOKENS / MTILE, NEXP), BLOCK, lds_bytes, stream>>>(
        x, W1, b1, W2, b2, lists, counts, yp);

    moe_combine<<<(TOKENS * DIM) / 256, 256, 0, stream>>>(yp, wbuf, out);
}
